// AttentionBlock_88313117540737
// MI455X (gfx1250) — compile-verified, matrix operand order checked by reading, not run on hardware
//
#include <hip/hip_runtime.h>

typedef __attribute__((ext_vector_type(16))) _Float16 v16h;
typedef __attribute__((ext_vector_type(8)))  _Float16 v8h;
typedef __attribute__((ext_vector_type(8)))  float    v8f;
typedef __attribute__((ext_vector_type(4)))  unsigned int v4u;
typedef __attribute__((ext_vector_type(8)))  int      v8si;
typedef __attribute__((ext_vector_type(4)))  int      v4si;

#define B_   32
#define C_   512
#define HW_  1024
#define G_   8
#define CPG_ 64
#define MTOT (B_ * HW_)   // 32768

// ---------------------------------------------------------------------------
// WMMA fragment loader: row-major [rows, ld] f16 source (global or LDS).
// A fragment (16x32, M x K): lane = (m, hi); per-lane data is two contiguous
// 16B chunks: K = [8*hi, 8*hi+8) and [16+8*hi, 16+8*hi+8).
// B fragment (32x16, K x N) mirrors it with N as the row index, so loading
// rows of an [N, K] row-major matrix with the same code yields D = A * B^T.
// ---------------------------------------------------------------------------
__device__ __forceinline__ v16h load_frag(const _Float16* base, int ld) {
    int lane = threadIdx.x & 31;
    int r  = lane & 15;
    int hi = lane >> 4;
    const _Float16* p = base + (size_t)r * ld + hi * 8;
    v8h lo = *(const v8h*)(p);
    v8h hv = *(const v8h*)(p + 16);
    v16h f;
#pragma unroll
    for (int i = 0; i < 8; ++i) { f[i] = lo[i]; f[8 + i] = hv[i]; }
    return f;
}

__device__ __forceinline__ v8f wmma_f16(v16h a, v16h b, v8f c) {
    return __builtin_amdgcn_wmma_f32_16x16x32_f16(false, a, false, b,
                                                  (short)0, c, false, false);
}

// ---------------------------------------------------------------------------
// Tensor Data Mover: 2-D tile load (global -> LDS), f16 elements.
// D# per CDNA5 ISA Sec. 8.3/8.4:
//   group0: [1:0]=count=1, [63:32]=lds byte addr, [120:64]=global byte addr,
//           [127:126]=type=2 ("image")
//   group1: [17:16]=data_size=1 (2 bytes), [79:48]=tensor_dim0,
//           [111:80]=tensor_dim1, [127:112]=tile_dim0, [143:128]=tile_dim1,
//           [207:160]=tensor_dim0_stride  (all dims in element units)
//   remaining groups: zero (tensor <= 2D)
// Issue from a single wave; sync with s_wait_tensorcnt + barrier.
// This toolchain carries the 6-arg builtin form (extra zero-filled group).
// ---------------------------------------------------------------------------
__device__ __forceinline__ void tdm_load_2d_f16(void* lds_dst, const void* gsrc,
                                                unsigned dim0, unsigned dim1,
                                                unsigned tile0, unsigned tile1,
                                                unsigned long long stride0) {
    unsigned long long ga = (unsigned long long)(uintptr_t)gsrc;
    unsigned lds_off = (unsigned)(uintptr_t)lds_dst;   // LDS aperture: addr[31:0]
    v4u g0;
    g0[0] = 1u;                                        // count=1, user D#
    g0[1] = lds_off;
    g0[2] = (unsigned)ga;
    g0[3] = (unsigned)(ga >> 32) | (2u << 30);         // type=2
    v8si g1;
    g1[0] = (int)(1u << 16);                           // data_size = 2 bytes
    g1[1] = (int)((dim0 & 0xFFFFu) << 16);
    g1[2] = (int)((dim0 >> 16) | ((dim1 & 0xFFFFu) << 16));
    g1[3] = (int)((dim1 >> 16) | (tile0 << 16));
    g1[4] = (int)(tile1 & 0xFFFFu);                    // tile_dim1 | tile_dim2=0
    g1[5] = (int)(unsigned)stride0;
    g1[6] = (int)(unsigned)((stride0 >> 32) & 0xFFFFu);
    g1[7] = 0;
    v4si z4 = {0, 0, 0, 0};
    v8si z8 = {0, 0, 0, 0, 0, 0, 0, 0};
    __builtin_amdgcn_tensor_load_to_lds(g0, g1, z4, z4, z8, 0);
}

// ---------------------------------------------------------------------------
// 1) GroupNorm statistics: one block per (batch, group); the 64 channels of a
//    group are contiguous (64*1024 floats) in [b,c,h,w] layout.
// ---------------------------------------------------------------------------
__global__ __launch_bounds__(256) void gn_stats(const float* __restrict__ x,
                                                float* __restrict__ mu,
                                                float* __restrict__ rstd) {
    int bg = blockIdx.x;                       // 0..255
    const float* p = x + (size_t)bg * (CPG_ * HW_);
    float s = 0.f, s2 = 0.f;
    for (int i = threadIdx.x; i < CPG_ * HW_; i += 256) {
        float v = p[i];
        s += v; s2 += v * v;
    }
    __shared__ float rs[256], rq[256];
    rs[threadIdx.x] = s; rq[threadIdx.x] = s2;
    __syncthreads();
    for (int off = 128; off > 0; off >>= 1) {
        if (threadIdx.x < off) {
            rs[threadIdx.x] += rs[threadIdx.x + off];
            rq[threadIdx.x] += rq[threadIdx.x + off];
        }
        __syncthreads();
    }
    if (threadIdx.x == 0) {
        const float inv_n = 1.0f / (float)(CPG_ * HW_);
        float m   = rs[0] * inv_n;
        float var = rq[0] * inv_n - m * m;
        mu[bg]   = m;
        rstd[bg] = rsqrtf(var + 1e-5f);
    }
}

// ---------------------------------------------------------------------------
// 2) Apply GN, convert to f16, transpose to [b, hw, c] (GEMM A layout).
// ---------------------------------------------------------------------------
__global__ __launch_bounds__(256) void gn_apply(const float* __restrict__ x,
                                                const float* __restrict__ gamma,
                                                const float* __restrict__ beta,
                                                const float* __restrict__ mu,
                                                const float* __restrict__ rstd,
                                                _Float16* __restrict__ xn) {
    size_t idx = (size_t)blockIdx.x * 256 + threadIdx.x;  // < 16777216
    int c  = (int)(idx & (C_ - 1));
    size_t m = idx >> 9;                                   // b*1024 + hw
    int b  = (int)(m >> 10);
    int hw = (int)(m & (HW_ - 1));
    int bg = b * G_ + (c >> 6);
    float v = x[((size_t)b * C_ + c) * HW_ + hw];
    float r = (v - mu[bg]) * rstd[bg] * gamma[c] + beta[c];
    xn[idx] = (_Float16)r;
}

// ---------------------------------------------------------------------------
// 3) Weights -> f16, keeping [c_out, c_in] row-major (B-fragment layout).
// ---------------------------------------------------------------------------
__global__ __launch_bounds__(256) void cvt_w(const float* __restrict__ qw,
                                             const float* __restrict__ kw,
                                             const float* __restrict__ vw,
                                             const float* __restrict__ pw,
                                             _Float16* __restrict__ w16) {
    int idx = blockIdx.x * 256 + threadIdx.x;      // < 4*262144
    int which = idx >> 18;
    int off   = idx & 262143;
    const float* src = (which == 0) ? qw : (which == 1) ? kw
                     : (which == 2) ? vw : pw;
    w16[idx] = (_Float16)src[off];
}

// ---------------------------------------------------------------------------
// 4) GEMM: out[m,n] = sum_k A[m,k] * W[n,k] + bias[n]   (M=32768, N=512, K=512)
//    Block 256 threads = 8 waves; block tile 128x128; wave tile 32x64.
//    The 128x512 f16 weight tile (128 KB) is TDM-staged into LDS once per
//    block, killing the 4x intra-block VMEM redundancy on B.
//    transposed==0 : out f16 [m, n]        ([b,hw,c])
//    transposed==1 : out f16 [b, n, hw]    (V stored channel-major for P*V)
// ---------------------------------------------------------------------------
#define GEMM_SMEM (128 * C_ * 2)   // 128 KB

__global__ __launch_bounds__(256) void gemm_nt_f16(const _Float16* __restrict__ A,
                                                   const _Float16* __restrict__ W,
                                                   const float* __restrict__ bias,
                                                   _Float16* __restrict__ out,
                                                   int transposed) {
    extern __shared__ char smem[];
    _Float16* Bs = (_Float16*)smem;            // [128][512]
    int tid = threadIdx.x, w = tid >> 5, lane = tid & 31;

    if (w == 0) {
        tdm_load_2d_f16(Bs, W + (size_t)blockIdx.y * 128 * C_,
                        C_, 128, C_, 128, C_);
        __builtin_amdgcn_s_wait_tensorcnt(0);
    }
    __syncthreads();

    int m0 = blockIdx.x * 128 + (w >> 1) * 32;
    int n0l = (w & 1) * 64;                    // local row in Bs

    v8f acc[2][4];
#pragma unroll
    for (int i = 0; i < 2; ++i)
#pragma unroll
        for (int j = 0; j < 4; ++j)
#pragma unroll
            for (int r = 0; r < 8; ++r) acc[i][j][r] = 0.f;

    for (int k0 = 0; k0 < C_; k0 += 32) {
        v16h a0 = load_frag(A + (size_t)m0 * C_ + k0, C_);
        v16h a1 = load_frag(A + (size_t)(m0 + 16) * C_ + k0, C_);
#pragma unroll
        for (int j = 0; j < 4; ++j) {
            v16h bf = load_frag(Bs + (size_t)(n0l + 16 * j) * C_ + k0, C_);
            acc[0][j] = wmma_f16(a0, bf, acc[0][j]);
            acc[1][j] = wmma_f16(a1, bf, acc[1][j]);
        }
    }

    int nc = lane & 15, hi = lane >> 4;
#pragma unroll
    for (int i = 0; i < 2; ++i)
#pragma unroll
        for (int j = 0; j < 4; ++j) {
            int n = blockIdx.y * 128 + n0l + 16 * j + nc;
            float bv = bias[n];
#pragma unroll
            for (int r = 0; r < 8; ++r) {
                int m = m0 + 16 * i + r + 8 * hi;
                float v = acc[i][j][r] + bv;
                if (!transposed) {
                    out[(size_t)m * C_ + n] = (_Float16)v;
                } else {
                    int b = m >> 10, hw = m & (HW_ - 1);
                    out[((size_t)b * C_ + n) * HW_ + hw] = (_Float16)v;
                }
            }
        }
}

// ---------------------------------------------------------------------------
// 5) Fused attention: one block per (batch, 16-row Q tile).
//    Q tile TDM-staged; S tile (16x1024 f32) in LDS via WMMA; softmax with
//    bank-conflict-free interleaved indexing; P f16 in LDS; then P*V.
// ---------------------------------------------------------------------------
#define ATT_SMEM (16 * 512 * 2 + 16 * 1024 * 4 + 16 * 1024 * 2 + 16 * 16 * 4)

__global__ __launch_bounds__(256) void attn_kernel(const _Float16* __restrict__ q16,
                                                   const _Float16* __restrict__ k16,
                                                   const _Float16* __restrict__ vt16,
                                                   _Float16* __restrict__ h16) {
    extern __shared__ char smem[];
    _Float16* Qs = (_Float16*)smem;                              // 16 x 512
    float*    Ss = (float*)(smem + 16384);                       // 16 x 1024
    _Float16* Ps = (_Float16*)(smem + 16384 + 65536);            // 16 x 1024
    float*    red = (float*)(smem + 16384 + 65536 + 32768);      // 16 x 16

    int tid = threadIdx.x;
    int b  = blockIdx.x >> 6;
    int qt = blockIdx.x & 63;
    int w = tid >> 5, lane = tid & 31, nc = lane & 15, hi = lane >> 4;

    // Stage the Q tile (16 rows x 512 ch = 8192 contiguous f16) via TDM.
    if (w == 0) {
        tdm_load_2d_f16(Qs, q16 + ((size_t)b * HW_ + qt * 16) * C_,
                        8192u, 1u, 8192u, 1u, 8192ull);
        __builtin_amdgcn_s_wait_tensorcnt(0);
    }
    __syncthreads();

    // ---- S = Q * K^T : each wave owns 8 column tiles of 16 ----
    for (int jj = 0; jj < 8; ++jj) {
        int jt = w * 8 + jj;
        v8f acc;
#pragma unroll
        for (int r = 0; r < 8; ++r) acc[r] = 0.f;
        const _Float16* krow = k16 + ((size_t)b * HW_ + jt * 16) * C_;
        for (int kk = 0; kk < C_; kk += 32) {
            v16h a  = load_frag(Qs + kk, C_);
            v16h bf = load_frag(krow + kk, C_);
            acc = wmma_f16(a, bf, acc);
        }
#pragma unroll
        for (int r = 0; r < 8; ++r)
            Ss[(size_t)(r + 8 * hi) * HW_ + jt * 16 + nc] = acc[r];
    }
    __syncthreads();

    // ---- softmax per row: 16 threads/row, interleaved (bank-conflict-free) --
    const float scale = 0.0441941738241592f;   // 1/sqrt(512)
    int row = tid >> 4, cs = tid & 15;
    float* srow = Ss + (size_t)row * HW_;
    float mx = -3.0e38f;
    for (int e = 0; e < 64; ++e) mx = fmaxf(mx, srow[cs + 16 * e]);
    red[row * 16 + cs] = mx;
    __syncthreads();
    float rmax = red[row * 16];
    for (int i = 1; i < 16; ++i) rmax = fmaxf(rmax, red[row * 16 + i]);
    __syncthreads();
    float sum = 0.f;
    for (int e = 0; e < 64; ++e) {
        int ix = cs + 16 * e;
        float p = __expf((srow[ix] - rmax) * scale);
        srow[ix] = p;
        sum += p;
    }
    red[row * 16 + cs] = sum;
    __syncthreads();
    float rsum = 0.f;
    for (int i = 0; i < 16; ++i) rsum += red[row * 16 + i];
    float inv = 1.0f / rsum;
    for (int e = 0; e < 64; ++e) {
        int ix = cs + 16 * e;
        Ps[(size_t)row * HW_ + ix] = (_Float16)(srow[ix] * inv);
    }
    __syncthreads();

    // ---- H = P * V : each wave owns 4 channel tiles of 16 (512/8/16) ----
    v8f acc2[4];
#pragma unroll
    for (int j = 0; j < 4; ++j)
#pragma unroll
        for (int r = 0; r < 8; ++r) acc2[j][r] = 0.f;

    for (int kk = 0; kk < HW_; kk += 32) {
        v16h a = load_frag(Ps + kk, HW_);
#pragma unroll
        for (int j = 0; j < 4; ++j) {
            const _Float16* vrow =
                vt16 + ((size_t)b * C_ + (w * 4 + j) * 16) * HW_ + kk;
            v16h bf = load_frag(vrow, HW_);
            acc2[j] = wmma_f16(a, bf, acc2[j]);
        }
    }
#pragma unroll
    for (int j = 0; j < 4; ++j)
#pragma unroll
        for (int r = 0; r < 8; ++r) {
            int m = qt * 16 + r + 8 * hi;
            int n = (w * 4 + j) * 16 + nc;
            h16[((size_t)b * HW_ + m) * C_ + n] = (_Float16)acc2[j][r];
        }
}

// ---------------------------------------------------------------------------
// 6) Projection GEMM + bias + residual, f32 output in [b, c, hw] layout.
//    Same TDM-staged weight tile as gemm_nt_f16.
// ---------------------------------------------------------------------------
__global__ __launch_bounds__(256) void gemm_proj(const _Float16* __restrict__ A,
                                                 const _Float16* __restrict__ W,
                                                 const float* __restrict__ bias,
                                                 const float* __restrict__ xres,
                                                 float* __restrict__ out) {
    extern __shared__ char smem[];
    _Float16* Bs = (_Float16*)smem;            // [128][512]
    int tid = threadIdx.x, w = tid >> 5, lane = tid & 31;

    if (w == 0) {
        tdm_load_2d_f16(Bs, W + (size_t)blockIdx.y * 128 * C_,
                        C_, 128, C_, 128, C_);
        __builtin_amdgcn_s_wait_tensorcnt(0);
    }
    __syncthreads();

    int m0 = blockIdx.x * 128 + (w >> 1) * 32;
    int n0l = (w & 1) * 64;

    v8f acc[2][4];
#pragma unroll
    for (int i = 0; i < 2; ++i)
#pragma unroll
        for (int j = 0; j < 4; ++j)
#pragma unroll
            for (int r = 0; r < 8; ++r) acc[i][j][r] = 0.f;

    for (int k0 = 0; k0 < C_; k0 += 32) {
        v16h a0 = load_frag(A + (size_t)m0 * C_ + k0, C_);
        v16h a1 = load_frag(A + (size_t)(m0 + 16) * C_ + k0, C_);
#pragma unroll
        for (int j = 0; j < 4; ++j) {
            v16h bf = load_frag(Bs + (size_t)(n0l + 16 * j) * C_ + k0, C_);
            acc[0][j] = wmma_f16(a0, bf, acc[0][j]);
            acc[1][j] = wmma_f16(a1, bf, acc[1][j]);
        }
    }

    int nc = lane & 15, hi = lane >> 4;
#pragma unroll
    for (int i = 0; i < 2; ++i)
#pragma unroll
        for (int j = 0; j < 4; ++j) {
            int n = blockIdx.y * 128 + n0l + 16 * j + nc;
            float bv = bias[n];
#pragma unroll
            for (int r = 0; r < 8; ++r) {
                int m = m0 + 16 * i + r + 8 * hi;
                int bb = m >> 10, hw = m & (HW_ - 1);
                size_t oi = ((size_t)bb * C_ + n) * HW_ + hw;
                out[oi] = xres[oi] + acc[i][j][r] + bv;
            }
        }
}

// ---------------------------------------------------------------------------
extern "C" void kernel_launch(void* const* d_in, const int* in_sizes, int n_in,
                              void* d_out, int out_size, void* d_ws, size_t ws_size,
                              hipStream_t stream) {
    const float* x     = (const float*)d_in[0];
    const float* gamma = (const float*)d_in[1];
    const float* beta  = (const float*)d_in[2];
    const float* q_w   = (const float*)d_in[3];
    const float* q_b   = (const float*)d_in[4];
    const float* k_w   = (const float*)d_in[5];
    const float* k_b   = (const float*)d_in[6];
    const float* v_w   = (const float*)d_in[7];
    const float* v_b   = (const float*)d_in[8];
    const float* p_w   = (const float*)d_in[9];
    const float* p_b   = (const float*)d_in[10];

    char* ws = (char*)d_ws;
    float*    mu   = (float*)ws;                 // 256 f32
    float*    rstd = (float*)(ws + 1024);        // 256 f32
    _Float16* w16  = (_Float16*)(ws + 4096);     // 4 * 512*512 f16
    size_t off = 4096 + (size_t)4 * C_ * C_ * 2;
    const size_t act = (size_t)MTOT * C_ * 2;    // 32 MB each
    _Float16* xn16 = (_Float16*)(ws + off); off += act;
    _Float16* q16  = (_Float16*)(ws + off); off += act;
    _Float16* k16  = (_Float16*)(ws + off); off += act;
    _Float16* vt16 = (_Float16*)(ws + off); off += act;
    _Float16* h16  = xn16;   // xn dead after QKV GEMMs -> reuse for attn output

    gn_stats<<<B_ * G_, 256, 0, stream>>>(x, mu, rstd);
    gn_apply<<<(MTOT * C_) / 256, 256, 0, stream>>>(x, gamma, beta, mu, rstd, xn16);
    cvt_w<<<(4 * C_ * C_) / 256, 256, 0, stream>>>(q_w, k_w, v_w, p_w, w16);

    dim3 gg(MTOT / 128, C_ / 128);
    gemm_nt_f16<<<gg, 256, GEMM_SMEM, stream>>>(xn16, w16,               q_b, q16, 0);
    gemm_nt_f16<<<gg, 256, GEMM_SMEM, stream>>>(xn16, w16 + C_ * C_,     k_b, k16, 0);
    gemm_nt_f16<<<gg, 256, GEMM_SMEM, stream>>>(xn16, w16 + 2 * C_ * C_, v_b, vt16, 1);

    attn_kernel<<<B_ * (HW_ / 16), 256, ATT_SMEM, stream>>>(q16, k16, vt16, h16);

    gemm_proj<<<gg, 256, GEMM_SMEM, stream>>>(h16, w16 + 3 * C_ * C_, p_b, x, (float*)d_out);
}
